// GraphAttention_82033875354135
// MI455X (gfx1250) — matrix operand order repaired and checked by reading, not compiled, not run on hardware
//
#include <hip/hip_runtime.h>

// ---------------------------------------------------------------------------
// GAT layer for MI455X (gfx1250, wave32, WMMA).
//   N=4096 nodes, FIN=256, H=8 heads, O=64 per head.
// Heavy math in v_wmma_f32_16x16x32_bf16 (f32 accumulate).
// Attention kernel double-buffers adj/f2 tiles in LDS via
// global_load_async_to_lds_* (ASYNCcnt) — CDNA5 async-copy path.
// ---------------------------------------------------------------------------

typedef __bf16 bf16_t;
typedef __attribute__((ext_vector_type(16))) __bf16 v16bf;
typedef __attribute__((ext_vector_type(8)))  __bf16 v8bf;
typedef __attribute__((ext_vector_type(8)))  float  v8f;

static constexpr int kN    = 4096;   // nodes
static constexpr int kFIN  = 256;    // input features
static constexpr int kH    = 8;      // heads
static constexpr int kO    = 64;     // out features per head
static constexpr int kHO   = kH * kO;
static constexpr int kJT   = 64;     // attention j-tile width

// ---------------------------------------------------------------------------
// Kernel 1: x (f32 [N][FIN]) -> xb (bf16 [N][FIN], row major)
// ---------------------------------------------------------------------------
__global__ void gat_pack_x(const float* __restrict__ x, bf16_t* __restrict__ xb) {
    int i = blockIdx.x * 256 + threadIdx.x;          // N*FIN threads
    xb[i] = (bf16_t)x[i];
}

// ---------------------------------------------------------------------------
// Kernel 2: weight (f32 [H][FIN][O]) -> wbT (bf16 [H][O][FIN])
// ---------------------------------------------------------------------------
__global__ void gat_pack_w(const float* __restrict__ w, bf16_t* __restrict__ wbT) {
    int i = blockIdx.x * 256 + threadIdx.x;          // H*FIN*O threads
    int h = i >> 14;                                  // / (FIN*O)
    int r = i & 16383;
    int f = r >> 6;                                   // / O
    int o = r & 63;
    wbT[(h * kO + o) * kFIN + f] = (bf16_t)w[i];
}

// ---------------------------------------------------------------------------
// Kernel 3: projection GEMM, one wave per (head, 16-row tile).
//   whT[h][o][n] (bf16) = sum_f xb[n][f] * wbT[h][o][f]
// ---------------------------------------------------------------------------
__global__ void __launch_bounds__(256)
gat_gemm1(const bf16_t* __restrict__ xb,
          const bf16_t* __restrict__ wbT,
          bf16_t* __restrict__ whT) {
    const int lane  = threadIdx.x & 31;
    const int wid   = (blockIdx.x << 3) + (threadIdx.x >> 5);  // 2048 waves
    const int h     = wid >> 8;                                 // 0..7
    const int mtile = wid & 255;                                // 0..255
    const int hv    = lane >> 4;                                // K-half
    const int irow  = lane & 15;

    const bf16_t* __restrict__ arow = xb + (size_t)(mtile * 16 + irow) * kFIN;
    const bf16_t* __restrict__ bcol = wbT + (size_t)(h * kO + irow) * kFIN;

    v8f acc[4] = {};
#pragma unroll
    for (int ks = 0; ks < 8; ++ks) {
        const int k0 = ks * 32;
        v8bf lo = *(const v8bf*)(arow + k0 + hv * 16);
        v8bf hi = *(const v8bf*)(arow + k0 + hv * 16 + 8);
        v16bf a = __builtin_shufflevector(lo, hi,
                    0,1,2,3,4,5,6,7,8,9,10,11,12,13,14,15);
#pragma unroll
        for (int nt = 0; nt < 4; ++nt) {
            v16bf b = *(const v16bf*)(bcol + nt * 16 * kFIN + k0 + hv * 16);
            acc[nt] = __builtin_amdgcn_wmma_f32_16x16x32_bf16(
                          false, a, false, b, (short)0, acc[nt], false, false);
        }
    }
#pragma unroll
    for (int nt = 0; nt < 4; ++nt) {
        v8bf st;
#pragma unroll
        for (int r = 0; r < 8; ++r) st[r] = (bf16_t)acc[nt][r];
        *(v8bf*)(whT + (size_t)(h * kO + nt * 16 + irow) * kN
                     + mtile * 16 + hv * 8) = st;
    }
}

// ---------------------------------------------------------------------------
// Kernel 4: f1[h][n] = sum_o whT[h][o][n]*attn_src[h][o]; f2 likewise.
// ---------------------------------------------------------------------------
__global__ void gat_f12(const bf16_t* __restrict__ whT,
                        const float* __restrict__ asrc,
                        const float* __restrict__ adst,
                        float* __restrict__ f1, float* __restrict__ f2) {
    int idx = blockIdx.x * 256 + threadIdx.x;        // H*N threads
    int h = idx >> 12;                                // / N
    int n = idx & 4095;
    const bf16_t* __restrict__ base = whT + (size_t)h * kO * kN + n;
    float s1 = 0.f, s2 = 0.f;
#pragma unroll 8
    for (int o = 0; o < kO; ++o) {
        float wv = (float)base[(size_t)o * kN];
        s1 += wv * asrc[h * kO + o];
        s2 += wv * adst[h * kO + o];
    }
    f1[idx] = s1;
    f2[idx] = s2;
}

// ---------------------------------------------------------------------------
// Kernel 5: fused attention + aggregation.
// Workgroup = 8 waves (1 per head), one 16-row i-block. adj streamed in
// 16x64 tiles via async global->LDS copies (read ONCE, shared by all heads),
// double-buffered with a single barrier per tile. Each wave builds the bf16
// A-fragment of P = exp(leaky(f1_i+f2_j))*mask in WMMA A layout and
// accumulates P @ wh with v_wmma_f32_16x16x32_bf16. Unnormalized-exp
// softmax: masked entries are exactly 0, divide by row sum at the end.
// ---------------------------------------------------------------------------
__global__ void __launch_bounds__(256)
gat_attn(const int* __restrict__ adj,
         const bf16_t* __restrict__ whT,
         const float* __restrict__ f1,
         const float* __restrict__ f2,
         const float* __restrict__ bias,
         float* __restrict__ out) {
    __shared__ int   adjT[2][16 * kJT];   // 2 x 4 KB
    __shared__ float f2s[2][kH][kJT];     // 2 x 2 KB

    const int lane = threadIdx.x & 31;
    const int h    = threadIdx.x >> 5;    // head == wave id in block
    const int hv   = lane >> 4;           // K-half / row-half
    const int irow = lane & 15;
    const int i0   = blockIdx.x * 16;

    // adj staging: 1024 ints / 256 threads = one int4 per thread
    const int srow = threadIdx.x >> 4;         // 0..15
    const int scol = (threadIdx.x & 15) * 4;   // 0,4,..,60

    const float  f1v = f1[h * kN + i0 + irow];
    const float* __restrict__ f2H = f2 + h * kN;
    const bf16_t* __restrict__ whH = whT + (size_t)h * kO * kN;

    v8f   acc[4] = {};
    float rs = 0.f;

    // Async stage of one 16x64 adj tile + 8x64 f2 chunk into LDS buffer `buf`.
    auto stage = [&](int j0, int buf) {
        unsigned aDst = (unsigned)(size_t)&adjT[buf][srow * kJT + scol];
        const int* aSrc = adj + (size_t)(i0 + srow) * kN + j0 + scol;
        asm volatile("global_load_async_to_lds_b128 %0, %1, off"
                     :: "v"(aDst), "v"(aSrc) : "memory");
        unsigned fDst = (unsigned)(size_t)&f2s[buf][h][lane * 2];
        const float* fSrc = f2H + j0 + lane * 2;
        asm volatile("global_load_async_to_lds_b64 %0, %1, off"
                     :: "v"(fDst), "v"(fSrc) : "memory");
    };

    stage(0, 0);
    asm volatile("s_wait_asynccnt 0x0" ::: "memory");
    __syncthreads();

    const int nTiles = kN / kJT;          // 64 tiles
    for (int jt = 0; jt < nTiles; ++jt) {
        const int cur = jt & 1;
        if (jt + 1 < nTiles) stage((jt + 1) * kJT, cur ^ 1);  // prefetch next

#pragma unroll
        for (int ks = 0; ks < 2; ++ks) {
            const int kb = ks * 32;
            // Vectorized LDS reads (ds_load_b128):
            //   flo/mlo <-> A elems 0..7  (K = kb + hv*16 + c)
            //   fhi/mhi <-> A elems 8..15 (K = kb + hv*16 + 8 + c)
            const float4* fv = (const float4*)&f2s[cur][h][kb];
            const float4  fA = fv[hv * 4 + 0], fB = fv[hv * 4 + 1];
            const float4  fC = fv[hv * 4 + 2], fD = fv[hv * 4 + 3];
            const int4* mv = (const int4*)&adjT[cur][irow * kJT + kb];
            const int4  mA = mv[hv * 4 + 0], mB = mv[hv * 4 + 1];
            const int4  mC = mv[hv * 4 + 2], mD = mv[hv * 4 + 3];

            const float flo[8] = {fA.x, fA.y, fA.z, fA.w, fB.x, fB.y, fB.z, fB.w};
            const float fhi[8] = {fC.x, fC.y, fC.z, fC.w, fD.x, fD.y, fD.z, fD.w};
            const int   mlo[8] = {mA.x, mA.y, mA.z, mA.w, mB.x, mB.y, mB.z, mB.w};
            const int   mhi[8] = {mC.x, mC.y, mC.z, mC.w, mD.x, mD.y, mD.z, mD.w};

            v16bf a;
#pragma unroll
            for (int c = 0; c < 8; ++c) {
                float e1 = f1v + flo[c];
                e1 = (e1 >= 0.f) ? e1 : 0.2f * e1;
                float p1 = (mlo[c] > 0) ? __expf(e1) : 0.f;
                bf16_t b1 = (bf16_t)p1;
                rs += (float)b1;              // sum exactly what the WMMA sees
                a[c] = b1;

                float e2 = f1v + fhi[c];
                e2 = (e2 >= 0.f) ? e2 : 0.2f * e2;
                float p2 = (mhi[c] > 0) ? __expf(e2) : 0.f;
                bf16_t b2 = (bf16_t)p2;
                rs += (float)b2;
                a[c + 8] = b2;
            }

#pragma unroll
            for (int nt = 0; nt < 4; ++nt) {
                v16bf b = *(const v16bf*)(whH + (size_t)(nt * 16 + irow) * kN
                                              + jt * kJT + kb + hv * 16);
                acc[nt] = __builtin_amdgcn_wmma_f32_16x16x32_bf16(
                              false, a, false, b, (short)0, acc[nt], false, false);
            }
        }

        // Next tile's async copies must have landed before anyone reads them.
        asm volatile("s_wait_asynccnt 0x0" ::: "memory");
        __syncthreads();
    }

    // Row sums: lanes L and L+16 hold partials of the same row (L&15).
    const float tot = rs + __shfl_xor(rs, 16);
    float sR[8];
#pragma unroll
    for (int r = 0; r < 8; ++r)
        sR[r] = __shfl(tot, hv * 8 + r);   // lane (hv*8+r) holds row hv*8+r

#pragma unroll
    for (int nt = 0; nt < 4; ++nt) {
        const float bv = bias[h * kO + nt * 16 + irow];
#pragma unroll
        for (int r = 0; r < 8; ++r) {
            const int m = i0 + hv * 8 + r;
            out[(size_t)m * kHO + h * kO + nt * 16 + irow] = acc[nt][r] / sR[r] + bv;
        }
    }
}

// ---------------------------------------------------------------------------
// Host-side launch
// ---------------------------------------------------------------------------
extern "C" void kernel_launch(void* const* d_in, const int* in_sizes, int n_in,
                              void* d_out, int out_size, void* d_ws, size_t ws_size,
                              hipStream_t stream) {
    (void)in_sizes; (void)n_in; (void)out_size; (void)ws_size;
    const float* x    = (const float*)d_in[0];
    const int*   adj  = (const int*)d_in[1];
    const float* w    = (const float*)d_in[2];
    const float* asrc = (const float*)d_in[3];
    const float* adst = (const float*)d_in[4];
    const float* bias = (const float*)d_in[5];
    float* out = (float*)d_out;

    char*  ws  = (char*)d_ws;
    size_t off = 0;
    auto carve = [&](size_t bytes) -> void* {
        void* p = ws + off;
        off += (bytes + 255) & ~(size_t)255;
        return p;
    };
    bf16_t* xb  = (bf16_t*)carve((size_t)kN * kFIN * sizeof(bf16_t));     // 2 MB
    bf16_t* wbT = (bf16_t*)carve((size_t)kH * kO * kFIN * sizeof(bf16_t));// 256 KB
    bf16_t* whT = (bf16_t*)carve((size_t)kH * kO * kN * sizeof(bf16_t));  // 4 MB
    float*  f1  = (float*) carve((size_t)kH * kN * sizeof(float));        // 128 KB
    float*  f2  = (float*) carve((size_t)kH * kN * sizeof(float));        // 128 KB

    gat_pack_x<<<(kN * kFIN) / 256, 256, 0, stream>>>(x, xb);
    gat_pack_w<<<(kH * kFIN * kO) / 256, 256, 0, stream>>>(w, wbT);
    gat_gemm1<<<(kH * (kN / 16)) / 8, 256, 0, stream>>>(xb, wbT, whT);
    gat_f12<<<(kH * kN) / 256, 256, 0, stream>>>(whT, asrc, adst, f1, f2);
    gat_attn<<<kN / 16, 256, 0, stream>>>(adj, whT, f1, f2, bias, out);
}
